// Attention_41858751267102
// MI455X (gfx1250) — matrix operand order repaired and checked by reading, not compiled, not run on hardware
//
#include <hip/hip_runtime.h>
#include <math.h>

typedef __bf16 bf16;
typedef __attribute__((ext_vector_type(16))) __bf16 bf16x16;
typedef __attribute__((ext_vector_type(8)))  __bf16 bf16x8;
typedef __attribute__((ext_vector_type(8)))  float  floatx8;
typedef int intx4 __attribute__((vector_size(4 * sizeof(int))));

#define D_MODEL 1024
#define N_Q     2048
#define N_KV    2048
#define N_HEADS 16
#define D_HEAD  64
#define R_REL   4095   // n + m - 1 relative positions
#define R_PAD   4096   // padded to multiple of 16 for WMMA tiling

#define AS_GLOBAL __attribute__((address_space(1)))
#define AS_LDS    __attribute__((address_space(3)))

#if __has_builtin(__builtin_amdgcn_global_load_async_to_lds_b128)
#define HAS_ASYNC_LDS 1
#else
#define HAS_ASYNC_LDS 0
#endif

union FragU { bf16x16 v; bf16x8 h[2]; };

// ---------------- WMMA fragment helpers (wave32, 16x16x32 bf16) -------------
// A 16x32: lane holds row=lane%16; K chunks at (lane/16)*8 and +16 (two 16B loads)
__device__ __forceinline__ bf16x16 load_fragA(const bf16* base, int ld, int row0, int k0) {
  int lane = threadIdx.x & 31;
  const bf16* p = base + (size_t)(row0 + (lane & 15)) * ld + k0 + ((lane >> 4) << 3);
  FragU f;
  f.h[0] = *reinterpret_cast<const bf16x8*>(p);
  f.h[1] = *reinterpret_cast<const bf16x8*>(p + 16);
  return f.v;
}
// B 32x16: lane holds col=lane%16; K chunks at (lane/16)*8 and +16 (two 16B loads)
__device__ __forceinline__ bf16x16 load_fragB(const bf16* base, int ld, int col0, int k0) {
  int lane = threadIdx.x & 31;
  const bf16* p = base + (size_t)(col0 + (lane & 15)) * ld + k0 + ((lane >> 4) << 3);
  FragU f;
  f.h[0] = *reinterpret_cast<const bf16x8*>(p);
  f.h[1] = *reinterpret_cast<const bf16x8*>(p + 16);
  return f.v;
}
__device__ __forceinline__ floatx8 wmma_bf16(bf16x16 a, bf16x16 b, floatx8 c) {
  return __builtin_amdgcn_wmma_f32_16x16x32_bf16(false, a, false, b, (short)0, c, false, false);
}

// 16-lane (row-group) reductions, wave32: xor masks 1,2,4,8 stay inside halves
__device__ __forceinline__ float row_max16(float v) {
#pragma unroll
  for (int m = 1; m < 16; m <<= 1) v = fmaxf(v, __shfl_xor(v, m, 32));
  return v;
}
__device__ __forceinline__ float row_sum16(float v) {
#pragma unroll
  for (int m = 1; m < 16; m <<= 1) v += __shfl_xor(v, m, 32);
  return v;
}

__device__ __forceinline__ void wait_async0() {
#if __has_builtin(__builtin_amdgcn_s_wait_asynccnt)
  __builtin_amdgcn_s_wait_asynccnt(0);
#else
  asm volatile("s_wait_asynccnt 0" ::: "memory");
#endif
}

// ---------------- prep kernels ---------------------------------------------
__global__ void k_f2bf(const float* __restrict__ in, bf16* __restrict__ out, int n) {
  int i = blockIdx.x * blockDim.x + threadIdx.x;
  if (i < n) out[i] = (bf16)in[i];
}

// to_{q,k,v}/for_pos_enc: (DQK,H,D) -> row-major [(h*64+d)][b]  (J x K for B-side)
__global__ void k_reorder_qkvw(const float* __restrict__ in, bf16* __restrict__ out) {
  int o = blockIdx.x * blockDim.x + threadIdx.x;
  if (o >= D_HEAD * N_HEADS * D_MODEL) return;
  int b = o & (D_MODEL - 1);
  int jd = o >> 10;          // h*64 + d
  int h = jd >> 6, d = jd & 63;
  out[o] = (bf16)in[((size_t)d * N_HEADS + h) * D_MODEL + b];
}

// to_out: (D, DV, H) -> row-major [dd][(h*64+c)]
__global__ void k_reorder_ow(const float* __restrict__ in, bf16* __restrict__ out) {
  int o = blockIdx.x * blockDim.x + threadIdx.x;
  if (o >= D_MODEL * N_HEADS * D_HEAD) return;
  int hc = o & (D_MODEL - 1);
  int dd = o >> 10;
  int h = hc >> 6, c = hc & 63;
  out[o] = (bf16)in[((size_t)dd * D_HEAD + c) * N_HEADS + h];
}

// sincos table, R_PAD x D_MODEL, row r -> position (r - (M-1)); row 4095 zeroed
__global__ void k_sincos(bf16* __restrict__ out) {
  int idx = blockIdx.x * blockDim.x + threadIdx.x;
  if (idx >= R_PAD * D_MODEL) return;
  int r = idx >> 10, c = idx & (D_MODEL - 1);
  float val = 0.f;
  if (r < R_REL) {
    float p = (float)(r - (N_KV - 1));
    int j = (c < 512) ? c : (c - 512);
    float invf = __expf(-((float)(2 * j) / (float)D_MODEL) * 9.210340371976184f); // ln 1e4
    float ph = p * invf;
    val = (c < 512) ? __sinf(ph) : __cosf(ph);
  }
  out[idx] = (bf16)val;
}

// ------- bf16 WMMA GEMM, 16x64 tile per wave: C(I x J) = A(I x K) * Bw(J x K)^T
template <int KD, bool OUT_F32>
__global__ __launch_bounds__(32, 1) void k_gemm(const bf16* __restrict__ A,
                                                const bf16* __restrict__ Bw,
                                                void* __restrict__ C, int ldc) {
  int lane = threadIdx.x & 31;
  int row0 = blockIdx.x << 4;
  int col0 = blockIdx.y << 6;  // 64 output columns per wave
  floatx8 acc[4];
#pragma unroll
  for (int t = 0; t < 4; t++)
#pragma unroll
    for (int i = 0; i < 8; i++) acc[t][i] = 0.f;

  const bf16* apre = A + (size_t)(row0 + (lane & 15)) * KD;
#pragma unroll 2
  for (int k0 = 0; k0 < KD; k0 += 32) {
    __builtin_prefetch(apre + k0 + 64, 0, 1);  // global_prefetch_b8 ahead on A
    bf16x16 a = load_fragA(A, KD, row0, k0);
#pragma unroll
    for (int t = 0; t < 4; t++) {
      bf16x16 b = load_fragB(Bw, KD, col0 + (t << 4), k0);
      acc[t] = wmma_bf16(a, b, acc[t]);
    }
  }
  int rb = row0 + ((lane >> 4) << 3);
#pragma unroll
  for (int t = 0; t < 4; t++) {
    int col = col0 + (t << 4) + (lane & 15);
    if (OUT_F32) {
      float* c = (float*)C;
#pragma unroll
      for (int g = 0; g < 8; g++) c[(size_t)(rb + g) * ldc + col] = acc[t][g];
    } else {
      bf16* c = (bf16*)C;
#pragma unroll
      for (int g = 0; g < 8; g++) c[(size_t)(rb + g) * ldc + col] = (bf16)acc[t][g];
    }
  }
}

// ---------------- fused flash attention with relative position --------------
// grid (N_Q/16, N_HEADS), block 32 (one wave). 32 keys per inner iteration:
// full-K PV WMMAs, softmax reductions amortized over 2 score tiles.
__global__ __launch_bounds__(32, 1) void k_attn(const bf16* __restrict__ Qm,
                                                const bf16* __restrict__ Km,
                                                const bf16* __restrict__ Vm,
                                                const bf16* __restrict__ PEm,
                                                bf16* __restrict__ ctx) {
  int lane = threadIdx.x & 31;
  int hlf = lane >> 4;     // 0: rows 0-7 of C tiles, 1: rows 8-15
  int jcol = lane & 15;    // C-tile column / A-tile row
  int n0 = blockIdx.x << 4;
  int h = blockIdx.y;
  const bf16* Qh = Qm + h * D_HEAD;
  const bf16* Kh = Km + h * D_HEAD;
  const bf16* Vh = Vm + h * D_HEAD;
  const bf16* Ph = PEm + h * D_HEAD;

  __shared__ __align__(16) float s_pl[16 * 48];  // pl tile (48 relative rows)
  __shared__ __align__(16) bf16 s_p[16 * 32];    // probabilities (A-side of PV)
  __shared__ __align__(16) bf16 s_v[32 * 64];    // staged V block (32 keys)

  bf16x16 qa0 = load_fragA(Qh, D_MODEL, n0, 0);
  bf16x16 qa1 = load_fragA(Qh, D_MODEL, n0, 32);

  floatx8 acc[4];
#pragma unroll
  for (int t = 0; t < 4; t++)
#pragma unroll
    for (int g = 0; g < 8; g++) acc[t][g] = 0.f;
  float mi[8], li[8];
#pragma unroll
  for (int g = 0; g < 8; g++) { mi[g] = -INFINITY; li[g] = 0.f; }

  for (int m0 = 0; m0 <= n0 + 15; m0 += 32) {  // causal 32-key blocks
    // async stage of V block (32 x 64 bf16): lane copies one 128B row
    {
      const bf16* src = Vh + (size_t)(m0 + lane) * D_MODEL;
      bf16* dst = s_v + lane * 64;
#if HAS_ASYNC_LDS
#pragma unroll
      for (int t = 0; t < 8; t++)
        __builtin_amdgcn_global_load_async_to_lds_b128(
            (AS_GLOBAL intx4*)(src + t * 8), (AS_LDS intx4*)(dst + t * 8), 0, 0);
#else
#pragma unroll
      for (int t = 0; t < 8; t++)
        *reinterpret_cast<bf16x8*>(dst + t * 8) =
            *reinterpret_cast<const bf16x8*>(src + t * 8);
#endif
    }

    // content scores: two 16x16 tiles covering keys m0..m0+31
    floatx8 s1, s2;
#pragma unroll
    for (int g = 0; g < 8; g++) { s1[g] = 0.f; s2[g] = 0.f; }
    s1 = wmma_bf16(qa0, load_fragB(Kh, D_MODEL, m0, 0), s1);
    s1 = wmma_bf16(qa1, load_fragB(Kh, D_MODEL, m0, 32), s1);
    s2 = wmma_bf16(qa0, load_fragB(Kh, D_MODEL, m0 + 16, 0), s2);
    s2 = wmma_bf16(qa1, load_fragB(Kh, D_MODEL, m0 + 16, 32), s2);

    // position logits: pl over 48 relative rows rlo..rlo+47
    // r(i,j) = (n0+i)-(m0+j)+M-1 = rlo + (i - j + 32)
    int rlo = n0 - m0 + N_KV - 33;
#pragma unroll
    for (int c = 0; c < 3; c++) {
      floatx8 pt;
#pragma unroll
      for (int g = 0; g < 8; g++) pt[g] = 0.f;
      pt = wmma_bf16(qa0, load_fragB(Ph, D_MODEL, rlo + (c << 4), 0), pt);
      pt = wmma_bf16(qa1, load_fragB(Ph, D_MODEL, rlo + (c << 4), 32), pt);
#pragma unroll
      for (int g = 0; g < 8; g++)
        s_pl[((hlf << 3) + g) * 48 + (c << 4) + jcol] = pt[g];
    }
    __syncthreads();

    // gather shifted position, add, causal mask (valid iff j - i <= n0 - m0)
    int dd = n0 - m0;
#pragma unroll
    for (int g = 0; g < 8; g++) {
      int i = (hlf << 3) + g;
      float v1 = s1[g] + s_pl[i * 48 + (i - jcol + 32)];        // j = jcol
      float v2 = s2[g] + s_pl[i * 48 + (i - jcol + 16)];        // j = jcol+16
      if (jcol - i > dd) v1 = -INFINITY;
      if (jcol + 16 - i > dd) v2 = -INFINITY;
      s1[g] = v1;
      s2[g] = v2;
    }

    // online softmax over 32 keys (one reduction pass for both tiles)
#pragma unroll
    for (int g = 0; g < 8; g++) {
      int i = (hlf << 3) + g;
      float rmax = row_max16(fmaxf(s1[g], s2[g]));
      float mn = fmaxf(mi[g], rmax);
      float scale = __expf(mi[g] - mn);
      float p1 = __expf(s1[g] - mn);
      float p2 = __expf(s2[g] - mn);
      float rs = row_sum16(p1 + p2);
      li[g] = li[g] * scale + rs;
      mi[g] = mn;
#pragma unroll
      for (int t = 0; t < 4; t++) acc[t][g] *= scale;
      s_p[i * 32 + jcol] = (bf16)p1;
      s_p[i * 32 + 16 + jcol] = (bf16)p2;
    }

#if HAS_ASYNC_LDS
    wait_async0();  // V tile resident in LDS
#endif
    __syncthreads();

    // PV: P is a full 16x32 A-frag, V_blk is 32x64 -> 4 full-K WMMAs
    FragU pf;
    {
      int kb = hlf << 3;
#pragma unroll
      for (int t = 0; t < 8; t++) {
        pf.v[t] = s_p[jcol * 32 + kb + t];          // K = kb .. kb+7
        pf.v[t + 8] = s_p[jcol * 32 + kb + 16 + t]; // K = kb+16 .. kb+23
      }
    }
#pragma unroll
    for (int t4 = 0; t4 < 4; t4++) {
      FragU vf;
#pragma unroll
      for (int t = 0; t < 16; t++)
        vf.v[t] = s_v[((hlf << 3) + (t & 7) + ((t >> 3) << 4)) * 64 + (t4 << 4) + jcol]; // K=hlf*8+(t&7)+(t>>3)*16
      acc[t4] = wmma_bf16(pf.v, vf.v, acc[t4]);
    }
    __syncthreads();
  }

  // epilogue: ctx[n][h*64+d] = acc/l
#pragma unroll
  for (int t4 = 0; t4 < 4; t4++)
#pragma unroll
    for (int g = 0; g < 8; g++) {
      int row = n0 + (hlf << 3) + g;
      ctx[(size_t)row * D_MODEL + h * D_HEAD + (t4 << 4) + jcol] =
          (bf16)(acc[t4][g] / li[g]);
    }
}

// ---------------- driver ----------------------------------------------------
extern "C" void kernel_launch(void* const* d_in, const int* in_sizes, int n_in,
                              void* d_out, int out_size, void* d_ws, size_t ws_size,
                              hipStream_t stream) {
  (void)in_sizes; (void)n_in; (void)out_size; (void)ws_size;
  const float* x_q    = (const float*)d_in[0];
  const float* x_kv   = (const float*)d_in[1];
  const float* to_q   = (const float*)d_in[2];
  const float* to_k   = (const float*)d_in[3];
  const float* to_v   = (const float*)d_in[4];
  const float* to_out = (const float*)d_in[5];
  const float* w_pe   = (const float*)d_in[6];

  char* w = (char*)d_ws;
  size_t off = 0;
  auto take = [&](size_t elems) -> bf16* {
    bf16* p = (bf16*)(w + off);
    off += ((elems * sizeof(bf16)) + 255) & ~(size_t)255;
    return p;
  };
  bf16* xq_bf  = take((size_t)N_Q * D_MODEL);
  bf16* xkv_bf = take((size_t)N_KV * D_MODEL);
  bf16* wq_bf  = take((size_t)D_MODEL * D_MODEL);
  bf16* wk_bf  = take((size_t)D_MODEL * D_MODEL);
  bf16* wv_bf  = take((size_t)D_MODEL * D_MODEL);
  bf16* wpe_bf = take((size_t)D_MODEL * D_MODEL);
  bf16* wo_bf  = take((size_t)D_MODEL * D_MODEL);
  bf16* sc_bf  = take((size_t)R_PAD * D_MODEL);
  bf16* q_bf   = take((size_t)N_Q * D_MODEL);
  bf16* k_bf   = take((size_t)N_KV * D_MODEL);
  bf16* v_bf   = take((size_t)N_KV * D_MODEL);
  bf16* pe_bf  = take((size_t)R_PAD * D_MODEL);
  bf16* ctx_bf = take((size_t)N_Q * D_MODEL);

  const int thr = 256;
  k_f2bf<<<dim3((N_Q * D_MODEL + thr - 1) / thr), thr, 0, stream>>>(x_q, xq_bf, N_Q * D_MODEL);
  k_f2bf<<<dim3((N_KV * D_MODEL + thr - 1) / thr), thr, 0, stream>>>(x_kv, xkv_bf, N_KV * D_MODEL);
  int wn = D_HEAD * N_HEADS * D_MODEL;
  k_reorder_qkvw<<<dim3((wn + thr - 1) / thr), thr, 0, stream>>>(to_q, wq_bf);
  k_reorder_qkvw<<<dim3((wn + thr - 1) / thr), thr, 0, stream>>>(to_k, wk_bf);
  k_reorder_qkvw<<<dim3((wn + thr - 1) / thr), thr, 0, stream>>>(to_v, wv_bf);
  k_reorder_qkvw<<<dim3((wn + thr - 1) / thr), thr, 0, stream>>>(w_pe, wpe_bf);
  k_reorder_ow<<<dim3((wn + thr - 1) / thr), thr, 0, stream>>>(to_out, wo_bf);
  k_sincos<<<dim3((R_PAD * D_MODEL + thr - 1) / thr), thr, 0, stream>>>(sc_bf);

  dim3 blk(32);
  k_gemm<D_MODEL, false><<<dim3(N_Q / 16, D_MODEL / 64), blk, 0, stream>>>(xq_bf, wq_bf, q_bf, D_MODEL);
  k_gemm<D_MODEL, false><<<dim3(N_KV / 16, D_MODEL / 64), blk, 0, stream>>>(xkv_bf, wk_bf, k_bf, D_MODEL);
  k_gemm<D_MODEL, false><<<dim3(N_KV / 16, D_MODEL / 64), blk, 0, stream>>>(xkv_bf, wv_bf, v_bf, D_MODEL);
  k_gemm<D_MODEL, false><<<dim3(R_PAD / 16, D_MODEL / 64), blk, 0, stream>>>(sc_bf, wpe_bf, pe_bf, D_MODEL);

  k_attn<<<dim3(N_Q / 16, N_HEADS), blk, 0, stream>>>(q_bf, k_bf, v_bf, pe_bf, ctx_bf);

  k_gemm<D_MODEL, true><<<dim3(N_Q / 16, D_MODEL / 64), blk, 0, stream>>>(ctx_bf, wo_bf, d_out, D_MODEL);
}
